// MultiHeadAttentionRoPE_70944269796020
// MI455X (gfx1250) — compile-verified, matrix operand order checked by reading, not run on hardware
//
#include <hip/hip_runtime.h>
#include <hip/hip_bf16.h>

// ---------------------------------------------------------------------------
// MultiHeadAttention + RoPE for MI455X (gfx1250, wave32, WMMA bf16, TDM)
// B=4, L=2048, D_MODEL=1024, H=16, HD=64
// ---------------------------------------------------------------------------

typedef __bf16 bf16_t;
typedef __attribute__((ext_vector_type(16))) __bf16 v16bf;
typedef __attribute__((ext_vector_type(8)))  float  v8f;
typedef __attribute__((ext_vector_type(4)))  int    v4i;

#define B_SZ    4
#define L_SEQ   2048
#define D_MODEL 1024
#define HEADS   16
#define HDIM    64
#define M_ROWS  (B_SZ * L_SEQ)   // 8192

#ifndef __has_builtin
#define __has_builtin(x) 0
#endif

// gfx1250 async global->LDS copies (ASYNCcnt-tracked), guarded with fallback
#if defined(__HIP_DEVICE_COMPILE__) && __has_builtin(__builtin_amdgcn_global_load_async_to_lds_b128)
#define HAS_ASYNC_LDS 1
#else
#define HAS_ASYNC_LDS 0
#endif

#if HAS_ASYNC_LDS && __has_builtin(__builtin_amdgcn_s_wait_asynccnt)
#define HAS_WAIT_ASYNC_BUILTIN 1
#else
#define HAS_WAIT_ASYNC_BUILTIN 0
#endif

// gfx1250 Tensor Data Mover (TENSORcnt-tracked), 6-arg builtin on this toolchain
#if defined(__HIP_DEVICE_COMPILE__) && __has_builtin(__builtin_amdgcn_tensor_load_to_lds) && __has_builtin(__builtin_amdgcn_s_wait_tensorcnt)
#define HAS_TDM 1
#else
#define HAS_TDM 0
#endif

#if HAS_ASYNC_LDS
typedef __attribute__((address_space(1))) v4i* gv4i_p;
typedef __attribute__((address_space(3))) v4i* lv4i_p;
__device__ __forceinline__ void async_b128(const bf16_t* g, bf16_t* l) {
    __builtin_amdgcn_global_load_async_to_lds_b128((gv4i_p)g, (lv4i_p)l, 0, 0);
}
#endif

__device__ __forceinline__ void wait_async0() {
#if HAS_ASYNC_LDS
#if HAS_WAIT_ASYNC_BUILTIN
    __builtin_amdgcn_s_wait_asynccnt(0);
#else
    asm volatile("s_wait_asynccnt 0" ::: "memory");
#endif
#endif
}

#if HAS_TDM
typedef __attribute__((ext_vector_type(4))) unsigned u32x4;
typedef __attribute__((ext_vector_type(8))) int      i32x8;
typedef __attribute__((ext_vector_type(4))) int      i32x4;
typedef __attribute__((address_space(3))) bf16_t*    lbf16_p;

// 2D tensor tile load (bf16): 64x64 tile, row = 32 dwords, LDS pad 4 dwords/row
// -> effective LDS row stride 72 bf16 elements (matches KS below).
__device__ __forceinline__ void tdm_load_2d_64x64(const bf16_t* gsrc, bf16_t* ldst) {
    unsigned long long ga = (unsigned long long)(size_t)gsrc;
    unsigned lds_off = (unsigned)(size_t)(lbf16_p)ldst;    // AS3 ptrtoint = LDS byte offset
    u32x4 g0;
    g0[0] = 1u;                                            // count=1 (valid user D#)
    g0[1] = lds_off;                                       // lds_addr
    g0[2] = (unsigned)(ga & 0xFFFFFFFFu);                  // global_addr[31:0]
    g0[3] = (unsigned)((ga >> 32) & 0x01FFFFFFu) | (2u << 30); // addr[56:32], type=2
    i32x8 g1;
    g1[0] = (int)((1u << 16)    // data_size = 2 bytes
                | (1u << 20)    // pad_enable
                | (4u << 22)    // pad_interval: 32 dwords (one 64-elem bf16 row)
                | (3u << 25));  // pad_amount: 4 dwords (8 bf16 elems)
    g1[1] = (int)(64u << 16);   // atomic_barrier_addr=0, tensor_dim0 = 64 (lo16)
    g1[2] = (int)(64u << 16);   // tensor_dim0 hi16=0, tensor_dim1 = 64 (lo16)
    g1[3] = (int)(64u << 16);   // tensor_dim1 hi16=0, tile_dim0 = 64
    g1[4] = (int)(64u);         // tile_dim1 = 64, tile_dim2 = 0
    g1[5] = (int)(64u);         // tensor_dim0_stride = 64 (lo32)
    g1[6] = 0;                  // stride hi16 = 0, tensor_dim1_stride lo16 = 0
    g1[7] = 0;
    i32x4 z4 = {0, 0, 0, 0};
    i32x8 z8 = {0, 0, 0, 0, 0, 0, 0, 0};
    __builtin_amdgcn_tensor_load_to_lds(g0, g1, z4, z4, z8, 0);
}
#endif

union Frag16 { v16bf v; unsigned u[8]; };
union PackB  { __bf16 h[2]; unsigned u; };

// k offset inside a 16-bit WMMA A/B fragment for VGPR slot j and lane-half hl
__device__ __forceinline__ int kOff(int j, int hl) {
    return (j & 3) * 2 + (j >> 2) * 16 + hl * 8;
}

// ---------------------------------------------------------------------------
// f32 -> bf16 conversion (grid-stride)
// ---------------------------------------------------------------------------
__global__ void cvt_bf16_kernel(const float* __restrict__ x, bf16_t* __restrict__ y, int n) {
    int i = blockIdx.x * blockDim.x + threadIdx.x;
    int stride = gridDim.x * blockDim.x;
    for (; i < n; i += stride) y[i] = (bf16_t)x[i];
}

// ---------------------------------------------------------------------------
// Projection GEMM: Y[m,n] = sum_k A[m,k] * W[n,k] + bias[n]
//   MODE 0: write f32 to outF[m*N + n]                       (final O-proj)
//   MODE 1: RoPE + scatter bf16 to outB[B,H,L,HD]            (Q, K)
//   MODE 2: scatter bf16 to outB[B,H,L,HD]                   (V)
// Block: 256 thr (8 waves). Block tile 128x128, wave tile 32x64, K step 32.
// ---------------------------------------------------------------------------
#define BM 128
#define BN 128
#define BK 32
#define SA 40   // padded LDS stride (bf16): 80 B rows -> 16B-aligned, conflict-free

template <int MODE>
__global__ __launch_bounds__(256)
void proj_kernel(const bf16_t* __restrict__ A, const bf16_t* __restrict__ W,
                 const float* __restrict__ bias,
                 float* __restrict__ outF, bf16_t* __restrict__ outB,
                 int M, int N, int K)
{
    __shared__ bf16_t sA[BM * SA];
    __shared__ bf16_t sB[BN * SA];

    const int tid  = threadIdx.x;
    const int wid  = tid >> 5;
    const int lane = tid & 31;
    const int hl   = lane >> 4;
    const int l16  = lane & 15;

    const int mBase = blockIdx.x * BM;
    const int nBase = blockIdx.y * BN;
    const int waveM = (wid >> 1) * 32;  // 0,32,64,96
    const int waveN = (wid & 1) * 64;   // 0,64

    v8f c[2][4];
#pragma unroll
    for (int ti = 0; ti < 2; ++ti)
#pragma unroll
        for (int tj = 0; tj < 4; ++tj)
#pragma unroll
            for (int j = 0; j < 8; ++j) c[ti][tj][j] = 0.0f;

    const int rT = tid >> 1;            // 0..127  (row for both A and B tiles)
    const int cT = (tid & 1) * 16;      // 0,16

    for (int k0 = 0; k0 < K; k0 += BK) {
        const bf16_t* gA = A + (size_t)(mBase + rT) * K + k0 + cT;
        const bf16_t* gB = W + (size_t)(nBase + rT) * K + k0 + cT;
        bf16_t* lA = sA + rT * SA + cT;
        bf16_t* lB = sB + rT * SA + cT;
#if HAS_ASYNC_LDS
        async_b128(gA,     lA);
        async_b128(gA + 8, lA + 8);
        async_b128(gB,     lB);
        async_b128(gB + 8, lB + 8);
#else
        {
            const uint4* g = (const uint4*)gA;
            uint4 p0 = g[0];
            uint4 p1 = g[1];
            unsigned* dst = (unsigned*)lA;
            dst[0] = p0.x; dst[1] = p0.y; dst[2] = p0.z; dst[3] = p0.w;
            dst[4] = p1.x; dst[5] = p1.y; dst[6] = p1.z; dst[7] = p1.w;
        }
        {
            const uint4* g = (const uint4*)gB;
            uint4 p0 = g[0];
            uint4 p1 = g[1];
            unsigned* dst = (unsigned*)lB;
            dst[0] = p0.x; dst[1] = p0.y; dst[2] = p0.z; dst[3] = p0.w;
            dst[4] = p1.x; dst[5] = p1.y; dst[6] = p1.z; dst[7] = p1.w;
        }
#endif
        if (k0 + BK < K) {
            __builtin_prefetch(gA + BK, 0, 0);
            __builtin_prefetch(gB + BK, 0, 0);
        }
        wait_async0();
        __syncthreads();

        Frag16 a[2], b[4];
#pragma unroll
        for (int ti = 0; ti < 2; ++ti)
#pragma unroll
            for (int j = 0; j < 8; ++j)
                a[ti].u[j] = *(const unsigned*)(sA + (waveM + ti * 16 + l16) * SA + kOff(j, hl));
#pragma unroll
        for (int tj = 0; tj < 4; ++tj)
#pragma unroll
            for (int j = 0; j < 8; ++j)
                b[tj].u[j] = *(const unsigned*)(sB + (waveN + tj * 16 + l16) * SA + kOff(j, hl));

#pragma unroll
        for (int ti = 0; ti < 2; ++ti)
#pragma unroll
            for (int tj = 0; tj < 4; ++tj)
                c[ti][tj] = __builtin_amdgcn_wmma_f32_16x16x32_bf16(
                    false, a[ti].v, false, b[tj].v, (short)0, c[ti][tj], false, false);

        __syncthreads();
    }

    // ---- epilogue ----
#pragma unroll
    for (int ti = 0; ti < 2; ++ti) {
#pragma unroll
        for (int tj = 0; tj < 4; ++tj) {
            const int nG = nBase + waveN + tj * 16 + l16;
            const float bval = bias[nG];
#pragma unroll
            for (int j = 0; j < 8; ++j) {
                const int mG = mBase + waveM + ti * 16 + j + 8 * hl;
                float v = c[ti][tj][j] + bval;
                if (MODE == 0) {
                    outF[(size_t)mG * N + nG] = v;
                } else {
                    const int bidx = mG / L_SEQ;
                    const int l    = mG % L_SEQ;
                    const int hh   = nG >> 6;
                    const int hd   = nG & 63;
                    if (MODE == 1) {
                        float partner = __shfl_xor(v, 1, 32);
                        float ang = (float)l * __powf(10000.0f, -(float)(hd & ~1) * (1.0f / 64.0f));
                        float cs = __cosf(ang), si = __sinf(ang);
                        v = (hd & 1) ? (partner * si + v * cs)
                                     : (v * cs - partner * si);
                    }
                    outB[(((size_t)bidx * HEADS + hh) * L_SEQ + l) * HDIM + hd] = (bf16_t)v;
                }
            }
        }
    }
}

// ---------------------------------------------------------------------------
// Flash attention (causal): per block = one (b,h) and one 64-row q tile.
// 4 waves, each wave owns 16 query rows; K/V staged in LDS by the Tensor Data
// Mover (async-copy / manual fallback); online softmax; P bounced through
// per-wave LDS to re-shape C-layout -> A-fragment layout.
// ---------------------------------------------------------------------------
#define KS 72   // padded LDS stride (bf16): 144 B rows; TDM pad reproduces this

__global__ __launch_bounds__(128)
void attn_kernel(const bf16_t* __restrict__ Q, const bf16_t* __restrict__ Kt,
                 const bf16_t* __restrict__ Vt, bf16_t* __restrict__ Oa)
{
    __shared__ bf16_t sK[64 * KS];
    __shared__ bf16_t sV[64 * KS];
    __shared__ bf16_t sP[4][16 * KS];

    const int tid  = threadIdx.x;
    const int wid  = tid >> 5;
    const int lane = tid & 31;
    const int hl   = lane >> 4;
    const int l16  = lane & 15;

    const int qt = blockIdx.x;           // q tile (64 rows)
    const int bh = blockIdx.y;           // b*H + h
    const int b  = bh >> 4;
    const int h  = bh & 15;
    const size_t baseBH = (size_t)bh * L_SEQ * HDIM;
    const int qRow0 = qt * 64 + wid * 16;      // wave's absolute first q row

    // Q fragments (two K-steps over HD=64), loaded once from global
    Frag16 aq[2];
    const bf16_t* qrow = Q + baseBH + (size_t)(qRow0 + l16) * HDIM;
#pragma unroll
    for (int st = 0; st < 2; ++st)
#pragma unroll
        for (int j = 0; j < 8; ++j)
            aq[st].u[j] = *(const unsigned*)(qrow + st * 32 + kOff(j, hl));

    float mrow[8], lrow[8];
    v8f o[4];
#pragma unroll
    for (int j = 0; j < 8; ++j) { mrow[j] = -1e30f; lrow[j] = 0.0f; }
#pragma unroll
    for (int t = 0; t < 4; ++t)
#pragma unroll
        for (int j = 0; j < 8; ++j) o[t][j] = 0.0f;

    for (int kt = 0; kt <= qt; ++kt) {
        const int kBase = kt * 64;

        // ---- stage K and V tiles (64x64 bf16 each) ----
#if HAS_TDM
        if (wid == 0) {
            tdm_load_2d_64x64(Kt + baseBH + (size_t)kBase * HDIM, sK);
            tdm_load_2d_64x64(Vt + baseBH + (size_t)kBase * HDIM, sV);
            __builtin_amdgcn_s_wait_tensorcnt(0);
        }
        __syncthreads();
#else
#pragma unroll
        for (int i = 0; i < 4; ++i) {
            int idx = tid * 4 + i;             // 0..511
            int r  = idx >> 3;
            int cc = (idx & 7) * 8;
            const bf16_t* gK = Kt + baseBH + (size_t)(kBase + r) * HDIM + cc;
            const bf16_t* gV = Vt + baseBH + (size_t)(kBase + r) * HDIM + cc;
#if HAS_ASYNC_LDS
            async_b128(gK, sK + r * KS + cc);
            async_b128(gV, sV + r * KS + cc);
#else
            uint4 pk = *(const uint4*)gK;
            uint4 pv = *(const uint4*)gV;
            unsigned* dk = (unsigned*)(sK + r * KS + cc);
            dk[0] = pk.x; dk[1] = pk.y; dk[2] = pk.z; dk[3] = pk.w;
            unsigned* dv = (unsigned*)(sV + r * KS + cc);
            dv[0] = pv.x; dv[1] = pv.y; dv[2] = pv.z; dv[3] = pv.w;
#endif
        }
        wait_async0();
        __syncthreads();
#endif

        // ---- S = Q K^T (wave: 16 x 64) ----
        float s[4][8];
#pragma unroll
        for (int tc = 0; tc < 4; ++tc) {
            v8f acc;
#pragma unroll
            for (int j = 0; j < 8; ++j) acc[j] = 0.0f;
#pragma unroll
            for (int st = 0; st < 2; ++st) {
                Frag16 bk;
#pragma unroll
                for (int j = 0; j < 8; ++j)
                    bk.u[j] = *(const unsigned*)(sK + (tc * 16 + l16) * KS + st * 32 + kOff(j, hl));
                acc = __builtin_amdgcn_wmma_f32_16x16x32_bf16(
                    false, aq[st].v, false, bk.v, (short)0, acc, false, false);
            }
#pragma unroll
            for (int j = 0; j < 8; ++j) s[tc][j] = acc[j];
        }

        // ---- scale + causal mask ----
#pragma unroll
        for (int tc = 0; tc < 4; ++tc) {
            const int kj = kBase + tc * 16 + l16;
#pragma unroll
            for (int j = 0; j < 8; ++j) {
                const int qi = qRow0 + j + 8 * hl;
                float v = s[tc][j] * 0.125f;       // 1/sqrt(64)
                s[tc][j] = (kj > qi) ? -1e30f : v;
            }
        }

        // ---- online softmax update (row reductions across 16-lane halves) ----
#pragma unroll
        for (int j = 0; j < 8; ++j) {
            float rm = s[0][j];
#pragma unroll
            for (int tc = 1; tc < 4; ++tc) rm = fmaxf(rm, s[tc][j]);
            rm = fmaxf(rm, __shfl_xor(rm, 1, 32));
            rm = fmaxf(rm, __shfl_xor(rm, 2, 32));
            rm = fmaxf(rm, __shfl_xor(rm, 4, 32));
            rm = fmaxf(rm, __shfl_xor(rm, 8, 32));
            const float mn = fmaxf(mrow[j], rm);
            const float alpha = __expf(mrow[j] - mn);
            mrow[j] = mn;
            float rs = 0.0f;
#pragma unroll
            for (int tc = 0; tc < 4; ++tc) {
                float p = __expf(s[tc][j] - mn);
                s[tc][j] = p;
                rs += p;
            }
            rs += __shfl_xor(rs, 1, 32);
            rs += __shfl_xor(rs, 2, 32);
            rs += __shfl_xor(rs, 4, 32);
            rs += __shfl_xor(rs, 8, 32);
            lrow[j] = lrow[j] * alpha + rs;
#pragma unroll
            for (int t = 0; t < 4; ++t) o[t][j] *= alpha;
        }

        // ---- write P (C-layout) into per-wave LDS as bf16 ----
        bf16_t* pw = sP[wid];
#pragma unroll
        for (int tc = 0; tc < 4; ++tc)
#pragma unroll
            for (int j = 0; j < 8; ++j)
                pw[(j + 8 * hl) * KS + tc * 16 + l16] = (bf16_t)s[tc][j];

        // ---- re-read P as A fragments ----
        Frag16 ap[2];
#pragma unroll
        for (int st = 0; st < 2; ++st)
#pragma unroll
            for (int j = 0; j < 8; ++j)
                ap[st].u[j] = *(const unsigned*)(pw + l16 * KS + st * 32 + kOff(j, hl));

        // ---- O += P V ----
#pragma unroll
        for (int tc = 0; tc < 4; ++tc) {
#pragma unroll
            for (int st = 0; st < 2; ++st) {
                Frag16 bv;
#pragma unroll
                for (int j = 0; j < 8; ++j) {
                    const int k = st * 32 + kOff(j, hl);
                    const int hd = tc * 16 + l16;
                    PackB p;
                    p.h[0] = sV[k * KS + hd];
                    p.h[1] = sV[(k + 1) * KS + hd];
                    bv.u[j] = p.u;
                }
                o[tc] = __builtin_amdgcn_wmma_f32_16x16x32_bf16(
                    false, ap[st].v, false, bv.v, (short)0, o[tc], false, false);
            }
        }
        __syncthreads();   // protect sK/sV before next tile overwrite
    }

    // ---- normalize + write [B, L, H*HD] bf16 ----
#pragma unroll
    for (int j = 0; j < 8; ++j) {
        const float inv = 1.0f / lrow[j];
        const int qi = qRow0 + j + 8 * hl;
#pragma unroll
        for (int t = 0; t < 4; ++t) {
            const int hd = t * 16 + l16;
            Oa[((size_t)b * L_SEQ + qi) * D_MODEL + h * HDIM + hd] = (bf16_t)(o[t][j] * inv);
        }
    }
}

// ---------------------------------------------------------------------------
// Launcher
// ---------------------------------------------------------------------------
extern "C" void kernel_launch(void* const* d_in, const int* in_sizes, int n_in,
                              void* d_out, int out_size, void* d_ws, size_t ws_size,
                              hipStream_t stream) {
    const float* query = (const float*)d_in[0];
    const float* Wq = (const float*)d_in[1];
    const float* bq = (const float*)d_in[2];
    const float* Wk = (const float*)d_in[3];
    const float* bk = (const float*)d_in[4];
    const float* Wv = (const float*)d_in[5];
    const float* bv = (const float*)d_in[6];
    const float* Wo = (const float*)d_in[7];
    const float* bo = (const float*)d_in[8];

    char* ws = (char*)d_ws;
    const size_t MB = 1024 * 1024;
    bf16_t* qx  = (bf16_t*)(ws + 0 * MB);    // 16 MB: query bf16 [8192,1024]
    bf16_t* Wqb = (bf16_t*)(ws + 16 * MB);   // 2 MB each
    bf16_t* Wkb = (bf16_t*)(ws + 18 * MB);
    bf16_t* Wvb = (bf16_t*)(ws + 20 * MB);
    bf16_t* Wob = (bf16_t*)(ws + 22 * MB);
    bf16_t* Qr  = (bf16_t*)(ws + 24 * MB);   // 16 MB: [B,H,L,HD]
    bf16_t* Kr  = (bf16_t*)(ws + 40 * MB);
    bf16_t* Vr  = (bf16_t*)(ws + 56 * MB);
    bf16_t* At  = (bf16_t*)(ws + 72 * MB);   // 16 MB: [B,L,D]

    const int nX = M_ROWS * D_MODEL;        // 8388608
    const int nW = D_MODEL * D_MODEL;       // 1048576

    cvt_bf16_kernel<<<2048, 256, 0, stream>>>(query, qx, nX);
    cvt_bf16_kernel<<<512, 256, 0, stream>>>(Wq, Wqb, nW);
    cvt_bf16_kernel<<<512, 256, 0, stream>>>(Wk, Wkb, nW);
    cvt_bf16_kernel<<<512, 256, 0, stream>>>(Wv, Wvb, nW);
    cvt_bf16_kernel<<<512, 256, 0, stream>>>(Wo, Wob, nW);

    dim3 gProj(M_ROWS / BM, D_MODEL / BN);
    proj_kernel<1><<<gProj, 256, 0, stream>>>(qx, Wqb, bq, nullptr, Qr, M_ROWS, D_MODEL, D_MODEL);
    proj_kernel<1><<<gProj, 256, 0, stream>>>(qx, Wkb, bk, nullptr, Kr, M_ROWS, D_MODEL, D_MODEL);
    proj_kernel<2><<<gProj, 256, 0, stream>>>(qx, Wvb, bv, nullptr, Vr, M_ROWS, D_MODEL, D_MODEL);

    dim3 gAttn(L_SEQ / 64, B_SZ * HEADS);
    attn_kernel<<<gAttn, 128, 0, stream>>>(Qr, Kr, Vr, At);

    proj_kernel<0><<<gProj, 256, 0, stream>>>(At, Wob, bo, (float*)d_out, nullptr, M_ROWS, D_MODEL, D_MODEL);
}
